// Set2Set_5574867550433
// MI455X (gfx1250) — matrix operand order repaired and checked by reading, not compiled, not run on hardware
//
#include <hip/hip_runtime.h>
#include <hip/hip_bf16.h>

// ---------------------------------------------------------------------------
// Set2Set (MEGNet-style) pooling for MI455X / gfx1250.
//  - GEMMs via v_wmma_f32_16x16x32_bf16 (bf16 in, f32 accumulate)
//  - B tiles staged with GLOBAL_LOAD_ASYNC_TO_LDS_B128 (ASYNCcnt) when available
//  - double-buffered LDS ping-pong to overlap staging with WMMA
//  - segment softmax fused single-pass per step using precomputed seg min/max
// ---------------------------------------------------------------------------

#define NA   131072   // atoms
#define FEAT 256      // input features == units
#define U    256      // units
#define NS   4096     // structures
#define KZ   768      // 2U + U (concat K for LSTM gemm)
#define NZ   1024     // 4U
#define STEPS 3

typedef __attribute__((ext_vector_type(16))) __bf16        v16bf;
typedef __attribute__((ext_vector_type(8)))  float         v8f;
typedef __attribute__((ext_vector_type(8)))  unsigned int  v8u;
typedef __attribute__((ext_vector_type(4)))  int           i4;

__device__ __forceinline__ unsigned short f32_to_bf16(float f) {
    unsigned int u = __builtin_bit_cast(unsigned int, f);
    u += 0x7fffu + ((u >> 16) & 1u);   // round-to-nearest-even
    return (unsigned short)(u >> 16);
}
__device__ __forceinline__ float sigmoid_f(float x) { return 1.0f / (1.0f + __expf(-x)); }
__device__ __forceinline__ float tanh_f(float x)    { return 2.0f / (1.0f + __expf(-2.0f * x)) - 1.0f; }

// --- CDNA5 async global->LDS copy (bypasses VGPRs, tracked by ASYNCcnt) -----
#if __has_builtin(__builtin_amdgcn_global_load_async_to_lds_b128)
#define HAVE_ASYNC_LDS 1
#else
#define HAVE_ASYNC_LDS 0
#endif

__device__ __forceinline__ void async_copy_b128(const unsigned short* g, unsigned short* l) {
#if HAVE_ASYNC_LDS
    // signature (from hipcc diagnostic): (int4 AS1*, int4 AS3*, imm offset, imm cpol)
    __builtin_amdgcn_global_load_async_to_lds_b128(
        (__attribute__((address_space(1))) i4*)(unsigned short*)g,
        (__attribute__((address_space(3))) i4*)l, 0, 0);
#else
    *(uint4*)l = *(const uint4*)g;
#endif
}
__device__ __forceinline__ void wait_async_lds() {
#if HAVE_ASYNC_LDS
#if __has_builtin(__builtin_amdgcn_s_wait_asynccnt)
    __builtin_amdgcn_s_wait_asynccnt(0);
#else
    asm volatile("s_wait_asynccnt 0x0" ::: "memory");
#endif
#endif
}

// ---------------------------------------------------------------------------
// Weight prep: cast + transpose to N-major bf16 so B fragments are K-contiguous
// ---------------------------------------------------------------------------
__global__ void cast_wd(const float* __restrict__ Wd, unsigned short* __restrict__ WdT) {
    int i = blockIdx.x * 256 + threadIdx.x;            // over U*FEAT
    if (i >= U * FEAT) return;
    int n = i / FEAT, k = i % FEAT;
    WdT[n * FEAT + k] = f32_to_bf16(Wd[k * U + n]);    // Wd: [FEAT, U]
}

__global__ void cast_wkr(const float* __restrict__ Wk, const float* __restrict__ Wr,
                         unsigned short* __restrict__ WkrT) {
    int i = blockIdx.x * 256 + threadIdx.x;            // over NZ*KZ
    if (i >= NZ * KZ) return;
    int n = i / KZ, k = i % KZ;
    float v = (k < 2 * U) ? Wk[k * NZ + n] : Wr[(k - 2 * U) * NZ + n];
    WkrT[n * KZ + k] = f32_to_bf16(v);
}

__global__ void zero_state(float* __restrict__ qstar, float* __restrict__ h,
                           float* __restrict__ c) {
    int i = blockIdx.x * 256 + threadIdx.x;            // NS*2U threads
    if (i < NS * 2 * U) qstar[i] = 0.0f;
    if (i < NS * U)     { h[i] = 0.0f; c[i] = 0.0f; }
}

// starts[s] = lower_bound(segment_ids, s); starts[NS] = NA  (ids are sorted)
__global__ void seg_starts(const int* __restrict__ seg, int* __restrict__ starts) {
    int s = blockIdx.x * 256 + threadIdx.x;
    if (s > NS) return;
    if (s == NS) { starts[s] = NA; return; }
    int lo = 0, hi = NA;
    while (lo < hi) { int mid = (lo + hi) >> 1; if (seg[mid] < s) lo = mid + 1; else hi = mid; }
    starts[s] = lo;
}

// ---------------------------------------------------------------------------
// BF16 WMMA GEMM:  C[M,N] = concat(A0[M,K0], A1[M,K1]) * B^T + bias
//   A0/A1: f32 row-major (cast to bf16 on LDS staging)
//   BT   : bf16, N-major [N][K]
// Block: 256 threads (8 waves), tile 128x128, BK=32. Wave sub-tile 32x64.
// Double-buffered LDS; B staged via async global->LDS b128.
// ---------------------------------------------------------------------------
#define BM 128
#define BN 128
#define BK 32
#define LDP 40   // padded LDS row stride (ushorts): conflict-free fragment reads

__global__ __launch_bounds__(256)
void gemm_bf16_wmma(const float* __restrict__ A0, int K0,
                    const float* __restrict__ A1, int K1,
                    const unsigned short* __restrict__ BT,
                    const float* __restrict__ bias,
                    float* __restrict__ C, int M, int N) {
    const int K  = K0 + K1;
    const int nk = K / BK;
    __shared__ unsigned short As[2][BM][LDP];   // [buf][m][k]
    __shared__ unsigned short Bs[2][BN][LDP];   // [buf][n][k]

    const int tid  = threadIdx.x;
    const int wave = tid >> 5;
    const int lane = tid & 31;
    const int bm   = blockIdx.y * BM;
    const int bn   = blockIdx.x * BN;
    const int wm   = (wave >> 1) * 32;       // 0,32,64,96
    const int wn   = (wave & 1) * 64;        // 0,64

    v8f acc[2][4] = {};

    const int r  = tid >> 1;                 // staging row (0..127)
    const int cb = (tid & 1) * 16;           // staging col base (0 or 16)

    const int mrow  = lane & 15;
    const int khalf = (lane >> 4) * 8;       // A frag: K split per ISA layout
    const int kb    = (lane >> 4) * 16;      // unused; B frag reads use khalf (same K order as A)

    auto stage = [&](int k0, int buf) {
        // A tile 128x32: f32 -> bf16 through VGPRs (conversion required)
        long m = bm + r;
        #pragma unroll
        for (int c2 = 0; c2 < 16; c2 += 4) {
            int kk = k0 + cb + c2;
            float4 v;
            if (kk < K0) v = *(const float4*)(&A0[m * K0 + kk]);
            else         v = *(const float4*)(&A1[m * K1 + (kk - K0)]);
            As[buf][r][cb + c2 + 0] = f32_to_bf16(v.x);
            As[buf][r][cb + c2 + 1] = f32_to_bf16(v.y);
            As[buf][r][cb + c2 + 2] = f32_to_bf16(v.z);
            As[buf][r][cb + c2 + 3] = f32_to_bf16(v.w);
        }
        // B tile 128x32: pure byte copy -> async global->LDS (no VGPR staging)
        const unsigned short* g = &BT[(long)(bn + r) * K + k0 + cb];
        async_copy_b128(g,     &Bs[buf][r][cb]);
        async_copy_b128(g + 8, &Bs[buf][r][cb + 8]);
    };

    stage(0, 0);
    for (int kt = 0; kt < nk; ++kt) {
        const int cur = kt & 1;
        wait_async_lds();        // my async B writes for tile kt have landed in LDS
        __syncthreads();         // everyone's staging visible
        if (kt + 1 < nk) stage((kt + 1) * BK, cur ^ 1);   // prefetch next tile

        // --- fragments (ISA 7.12.2 layouts) + WMMA on current buffer ---
        v16bf afrag[2];
        #pragma unroll
        for (int i = 0; i < 2; ++i) {
            const uint4* p0 = (const uint4*)(&As[cur][wm + i * 16 + mrow][khalf]);
            const uint4* p1 = (const uint4*)(&As[cur][wm + i * 16 + mrow][16 + khalf]);
            uint4 lo = p0[0], hi = p1[0];
            v8u u = { lo.x, lo.y, lo.z, lo.w, hi.x, hi.y, hi.z, hi.w };
            afrag[i] = __builtin_bit_cast(v16bf, u);
        }
        v16bf bfrag[4];
        #pragma unroll
        for (int j = 0; j < 4; ++j) {
            const uint4* p0 = (const uint4*)(&Bs[cur][wn + j * 16 + (lane & 15)][khalf]);
            const uint4* p1 = (const uint4*)(&Bs[cur][wn + j * 16 + (lane & 15)][16 + khalf]);
            uint4 lo = p0[0], hi = p1[0];
            v8u u = { lo.x, lo.y, lo.z, lo.w, hi.x, hi.y, hi.z, hi.w };
            bfrag[j] = __builtin_bit_cast(v16bf, u);
        }
        #pragma unroll
        for (int i = 0; i < 2; ++i)
            #pragma unroll
            for (int j = 0; j < 4; ++j)
                acc[i][j] = __builtin_amdgcn_wmma_f32_16x16x32_bf16(
                    false, afrag[i], false, bfrag[j], (short)0, acc[i][j], false, false);
    }

    // --- epilogue: D layout (VGPR r -> M = r + 8*(lane>=16); N = lane&15) ---
    #pragma unroll
    for (int i = 0; i < 2; ++i) {
        int mbase = bm + wm + i * 16 + ((lane >> 4) << 3);
        #pragma unroll
        for (int j = 0; j < 4; ++j) {
            int n = bn + wn + j * 16 + (lane & 15);
            float bv = bias[n];
            #pragma unroll
            for (int rr = 0; rr < 8; ++rr)
                C[(long)(mbase + rr) * N + n] = acc[i][j][rr] + bv;
        }
    }
}

// ---------------------------------------------------------------------------
// Per-segment per-channel min/max of x (step-invariant -> computed once)
// ---------------------------------------------------------------------------
__global__ void seg_minmax(const float* __restrict__ x, const int* __restrict__ starts,
                           float* __restrict__ smax, float* __restrict__ smin) {
    int s = blockIdx.x, ch = threadIdx.x;
    int b = starts[s], e = starts[s + 1];
    float mx = -3.4e38f, mn = 3.4e38f;
    for (int r = b; r < e; ++r) {
        float v = x[(long)r * U + ch];
        mx = fmaxf(mx, v);
        mn = fminf(mn, v);
    }
    smax[s * U + ch] = mx;
    smin[s * U + ch] = mn;
}

// ---------------------------------------------------------------------------
// LSTM gate nonlinearity; writes qt = h_new into first half of q_star
// ---------------------------------------------------------------------------
__global__ void lstm_elem(const float* __restrict__ z, float* __restrict__ h,
                          float* __restrict__ c, float* __restrict__ qstar) {
    int idx = blockIdx.x * 256 + threadIdx.x;   // NS*U
    int s = idx >> 8, u = idx & 255;
    const float* zr = z + (long)s * NZ;
    float ig = sigmoid_f(zr[u]);
    float fg = sigmoid_f(zr[U + u]);
    float gg = tanh_f(zr[2 * U + u]);
    float og = sigmoid_f(zr[3 * U + u]);
    float cn = fg * c[idx] + ig * gg;
    float hn = og * tanh_f(cn);
    c[idx] = cn;
    h[idx] = hn;
    qstar[(long)s * (2 * U) + u] = hn;
}

// ---------------------------------------------------------------------------
// Fused segment softmax + weighted readout (single pass over x per step):
//   m = qt>=0 ? qt*segmax : qt*segmin;  rt = sum(exp(x*qt-m)*x)/sum(exp(x*qt-m))
// ---------------------------------------------------------------------------
__global__ void set2set_attn(const float* __restrict__ x, const int* __restrict__ starts,
                             const float* __restrict__ h, const float* __restrict__ smax,
                             const float* __restrict__ smin, float* __restrict__ qstar) {
    int s = blockIdx.x, ch = threadIdx.x;
    int b = starts[s], e = starts[s + 1];
    float qt = h[s * U + ch];
    float m  = (qt >= 0.0f) ? qt * smax[s * U + ch] : qt * smin[s * U + ch];
    float ssum = 0.0f, rsum = 0.0f;
    for (int r = b; r < e; ++r) {
        float xv = x[(long)r * U + ch];
        float w  = __expf(xv * qt - m);
        ssum += w;
        rsum += w * xv;
    }
    float rt = (e > b) ? (rsum / ssum) : 0.0f;
    qstar[(long)s * (2 * U) + U + ch] = rt;
}

// ---------------------------------------------------------------------------
static inline size_t align_up(size_t v, size_t a) { return (v + a - 1) & ~(a - 1); }

extern "C" void kernel_launch(void* const* d_in, const int* in_sizes, int n_in,
                              void* d_out, int out_size, void* d_ws, size_t ws_size,
                              hipStream_t stream) {
    const float* field = (const float*)d_in[0];
    const float* Wd    = (const float*)d_in[1];
    const float* bd    = (const float*)d_in[2];
    const float* Wk    = (const float*)d_in[3];
    const float* Wr    = (const float*)d_in[4];
    const float* bl    = (const float*)d_in[5];
    const int*   seg   = (const int*)d_in[6];
    float* qstar = (float*)d_out;               // [NS, 2U] lives directly in d_out

    char* ws = (char*)d_ws;
    size_t off = 0;
    float*          x     = (float*)(ws + off); off = align_up(off + (size_t)NA * U * 4, 256);
    unsigned short* WdT   = (unsigned short*)(ws + off); off = align_up(off + (size_t)U * FEAT * 2, 256);
    unsigned short* WkrT  = (unsigned short*)(ws + off); off = align_up(off + (size_t)NZ * KZ * 2, 256);
    float*          z     = (float*)(ws + off); off = align_up(off + (size_t)NS * NZ * 4, 256);
    float*          h     = (float*)(ws + off); off = align_up(off + (size_t)NS * U * 4, 256);
    float*          c     = (float*)(ws + off); off = align_up(off + (size_t)NS * U * 4, 256);
    float*          smax  = (float*)(ws + off); off = align_up(off + (size_t)NS * U * 4, 256);
    float*          smin  = (float*)(ws + off); off = align_up(off + (size_t)NS * U * 4, 256);
    int*            starts= (int*)(ws + off);   off = align_up(off + (size_t)(NS + 1) * 4, 256);

    // prep
    cast_wd  <<<(U * FEAT + 255) / 256, 256, 0, stream>>>(Wd, WdT);
    cast_wkr <<<(NZ * KZ + 255) / 256, 256, 0, stream>>>(Wk, Wr, WkrT);
    zero_state<<<(NS * 2 * U + 255) / 256, 256, 0, stream>>>(qstar, h, c);
    seg_starts<<<(NS + 1 + 255) / 256, 256, 0, stream>>>(seg, starts);

    // x = field @ Wd + bd   (M=131072, N=256, K=256)
    gemm_bf16_wmma<<<dim3(U / BN, NA / BM), 256, 0, stream>>>(
        field, FEAT, nullptr, 0, WdT, bd, x, NA, U);

    // step-invariant per-segment min/max of x
    seg_minmax<<<NS, U, 0, stream>>>(x, starts, smax, smin);

    for (int t = 0; t < STEPS; ++t) {
        // z = [qstar, h] @ [Wk; Wr] + b   (M=4096, N=1024, K=768)
        gemm_bf16_wmma<<<dim3(NZ / BN, NS / BM), 256, 0, stream>>>(
            qstar, 2 * U, h, U, WkrT, bl, z, NS, NZ);
        lstm_elem<<<(NS * U + 255) / 256, 256, 0, stream>>>(z, h, c, qstar);
        set2set_attn<<<NS, U, 0, stream>>>(x, starts, h, smax, smin, qstar);
    }
    (void)in_sizes; (void)n_in; (void)out_size; (void)ws_size;
}
